// FaceClassifierDGL_15925738734017
// MI455X (gfx1250) — matrix operand order repaired and checked by reading, not compiled, not run on hardware
//
#include <hip/hip_runtime.h>
#include <math.h>

typedef __attribute__((ext_vector_type(16))) _Float16 v16h;
typedef __attribute__((ext_vector_type(8)))  _Float16 v8h;
typedef __attribute__((ext_vector_type(4)))  _Float16 v4h;
typedef __attribute__((ext_vector_type(8)))  float    v8f;

#define KNN 16

// ---------------------------------------------------------------------------
// 1) Brute-force kNN: each thread owns one dst node, keeps a sorted top-16
//    (ascending d2) in registers; candidate centers streamed through LDS.
// ---------------------------------------------------------------------------
__global__ __launch_bounds__(256) void knn_kernel(const float* __restrict__ centers,
                                                  int* __restrict__ idx, int N) {
  __shared__ float sx[256], sy[256], sz[256];
  const int i = blockIdx.x * 256 + threadIdx.x;
  float cx = 0.f, cy = 0.f, cz = 0.f;
  if (i < N) { cx = centers[3*i]; cy = centers[3*i+1]; cz = centers[3*i+2]; }

  float bd[KNN]; int bi[KNN];
#pragma unroll
  for (int t = 0; t < KNN; ++t) { bd[t] = 3.0e38f; bi[t] = 0; }

  for (int t0 = 0; t0 < N; t0 += 256) {
    __syncthreads();
    const int c = t0 + (int)threadIdx.x;
    if (c < N) {
      sx[threadIdx.x] = centers[3*c];
      sy[threadIdx.x] = centers[3*c+1];
      sz[threadIdx.x] = centers[3*c+2];
    }
    __syncthreads();
    int tl = N - t0; if (tl > 256) tl = 256;
    for (int j = 0; j < tl; ++j) {
      const float dx = cx - sx[j], dy = cy - sy[j], dz = cz - sz[j];
      const float d2 = dx*dx + dy*dy + dz*dz;
      if (d2 < bd[KNN-1]) {                 // strict < keeps earliest index on ties
        const int cand = t0 + j;
#pragma unroll
        for (int t = KNN-1; t >= 1; --t) {  // old values read before overwrite (descending)
          const bool gt = bd[t]   > d2;
          const bool sh = bd[t-1] > d2;
          const float nd = sh ? bd[t-1] : d2;
          const int   ni = sh ? bi[t-1] : cand;
          if (gt) { bd[t] = nd; bi[t] = ni; }
        }
        if (bd[0] > d2) { bd[0] = d2; bi[0] = cand; }
      }
    }
  }
  if (i < N) {
#pragma unroll
    for (int t = 0; t < KNN; ++t) idx[i*KNN + t] = bi[t];
  }
}

// ---------------------------------------------------------------------------
// 2) Degree / norm helpers.  norm folds dst_norm = 1/sqrt(K) = 0.25.
// ---------------------------------------------------------------------------
__global__ void zero_i32_kernel(int* __restrict__ p, int n) {
  const int i = blockIdx.x * blockDim.x + threadIdx.x;
  if (i < n) p[i] = 0;
}

__global__ void count_deg_kernel(const int* __restrict__ idx, int* __restrict__ deg, int nEdges) {
  const int t = blockIdx.x * blockDim.x + threadIdx.x;
  if (t < nEdges) atomicAdd(&deg[idx[t]], 1);
}

__global__ void norm_kernel(const int* __restrict__ deg, float* __restrict__ nrm,
                            int N, float dstn) {
  const int i = blockIdx.x * blockDim.x + threadIdx.x;
  if (i < N) {
    int d = deg[i]; if (d < 1) d = 1;
    nrm[i] = dstn * rsqrtf((float)d);
  }
}

// ---------------------------------------------------------------------------
// 3) Pack fp32 weight matrix W[F,H] into f16 B-fragment layout:
//    element ((ct*(F/32)+ks)*32 + lane)*16 + j  <-  W[ks*32 + (lane<16?0:16) + j][ct*16 + lane%16]
//    -> each lane's 16 halves are 32 contiguous bytes (two b128 loads).
// ---------------------------------------------------------------------------
__global__ void pack_w_kernel(const float* __restrict__ W, _Float16* __restrict__ Wp,
                              int F, int H) {
  const int t = blockIdx.x * blockDim.x + threadIdx.x;
  if (t >= F * H) return;
  const int j    = t & 15;
  const int lane = (t >> 4) & 31;
  const int rest = t >> 9;
  const int ksteps = F >> 5;
  const int ks = rest % ksteps;
  const int ct = rest / ksteps;
  const int n  = (ct << 4) + (lane & 15);
  const int k  = (ks << 5) + ((lane < 16) ? 0 : 16) + j;
  Wp[t] = (_Float16)W[k * H + n];
}

// ---------------------------------------------------------------------------
// 4) Neighborhood aggregation: agg[i,f] = sum_k nrm[idx[i,k]] * h[idx[i,k], f]
//    (nrm already includes dst_norm).  Output f16 (WMMA A operand feed).
// ---------------------------------------------------------------------------
__global__ __launch_bounds__(256) void aggregate_kernel(const float* __restrict__ h,
                                                        const int* __restrict__ idx,
                                                        const float* __restrict__ nrm,
                                                        _Float16* __restrict__ agg,
                                                        int N, int F) {
  const int chunks = F >> 2;
  const int t = blockIdx.x * blockDim.x + threadIdx.x;
  if (t >= N * chunks) return;
  const int i = t / chunks;
  const int f = (t - i * chunks) << 2;
  const int* __restrict__ row = idx + i * KNN;
  float a0 = 0.f, a1 = 0.f, a2 = 0.f, a3 = 0.f;
#pragma unroll 4
  for (int k = 0; k < KNN; ++k) {
    const int s = row[k];
    const float w = nrm[s];
    const float4 v = *(const float4*)(h + (size_t)s * F + f);
    a0 = fmaf(w, v.x, a0);
    a1 = fmaf(w, v.y, a1);
    a2 = fmaf(w, v.z, a2);
    a3 = fmaf(w, v.w, a3);
  }
  v4h o;
  o[0] = (_Float16)a0; o[1] = (_Float16)a1; o[2] = (_Float16)a2; o[3] = (_Float16)a3;
  *(v4h*)(agg + (size_t)i * F + f) = o;
}

// ---------------------------------------------------------------------------
// 5) WMMA GEMM:  out[N,H] = relu(A[N,F](f16) x W[F,H](packed f16) + bias)
//    One wave per 16x16 output tile; K consumed 32 at a time via
//    v_wmma_f32_16x16x32_f16.  Tile guard is wave-uniform (EXEC all-ones).
// ---------------------------------------------------------------------------
__global__ __launch_bounds__(256) void gemm_wmma_kernel(const _Float16* __restrict__ A,
                                                        const _Float16* __restrict__ Wp,
                                                        const float* __restrict__ bias,
                                                        float* __restrict__ out,
                                                        int Nrows, int F, int H, int do_relu) {
  const int lane = threadIdx.x & 31;
  const int wave = threadIdx.x >> 5;
  const int colTiles = H >> 4;
  const int rowTiles = (Nrows + 15) >> 4;
  const int tile = blockIdx.x * (blockDim.x >> 5) + wave;
  if (tile >= rowTiles * colTiles) return;   // wave-uniform
  const int rt = tile / colTiles;
  const int ct = tile - rt * colTiles;

  int row = (rt << 4) + (lane & 15);
  if (row >= Nrows) row = Nrows - 1;         // clamp pad rows to valid memory
  const int kb = (lane < 16) ? 0 : 16;       // contiguous A layout: lanes 0-15 K{0-15}, 16-31 K{16-31}
  const _Float16* __restrict__ Ab = A + (size_t)row * F + kb;
  const _Float16* __restrict__ Bb = Wp + (((size_t)ct * (size_t)(F >> 5)) << 9) + ((size_t)lane << 4);

  v8f c = {};
  const int ksteps = F >> 5;
  for (int ks = 0; ks < ksteps; ++ks) {
    union { v16h v; v8h h[2]; } a;
    a.h[0] = *(const v8h*)(Ab + (ks << 5));        // K = kb .. kb+7
    a.h[1] = *(const v8h*)(Ab + (ks << 5) + 8);    // K = kb+8 .. kb+15
    const v16h b = *(const v16h*)(Bb + ((size_t)ks << 9));
    c = __builtin_amdgcn_wmma_f32_16x16x32_f16(false, a.v, false, b,
                                               (short)0, c, false, false);
  }

  const int n = (ct << 4) + (lane & 15);
  const float bb = bias[n];
  const int mbase = (rt << 4) + ((lane >> 4) << 3);  // C/D layout: lanes 0-15 M=r, 16-31 M=8+r
#pragma unroll
  for (int r = 0; r < 8; ++r) {
    const int m = mbase + r;
    if (m < Nrows) {
      float v = c[r] + bb;
      if (do_relu) v = fmaxf(v, 0.0f);
      out[(size_t)m * H + n] = v;
    }
  }
}

// ---------------------------------------------------------------------------
// 6) Final classifier: sigmoid(h @ Wf + bf)
// ---------------------------------------------------------------------------
__global__ void final_kernel(const float* __restrict__ h, const float* __restrict__ Wf,
                             const float* __restrict__ bf, float* __restrict__ out,
                             int N, int H) {
  const int i = blockIdx.x * blockDim.x + threadIdx.x;
  if (i >= N) return;
  const float* __restrict__ row = h + (size_t)i * H;
  float z = bf[0];
#pragma unroll 8
  for (int f = 0; f < H; ++f) z = fmaf(row[f], Wf[f], z);
  out[i] = 1.0f / (1.0f + expf(-z));
}

// ---------------------------------------------------------------------------
// Host launcher
// ---------------------------------------------------------------------------
static inline int cdiv_host(int a, int b) { return (a + b - 1) / b; }

extern "C" void kernel_launch(void* const* d_in, const int* in_sizes, int n_in,
                              void* d_out, int out_size, void* d_ws, size_t ws_size,
                              hipStream_t stream) {
  const float* x       = (const float*)d_in[0];
  const float* centers = (const float*)d_in[1];
  const float* W0      = (const float*)d_in[2];
  const float* b0      = (const float*)d_in[3];
  const float* W1      = (const float*)d_in[4];
  const float* b1      = (const float*)d_in[5];
  const float* W2      = (const float*)d_in[6];
  const float* b2      = (const float*)d_in[7];
  const float* Wf      = (const float*)d_in[8];
  const float* bf      = (const float*)d_in[9];
  float* out = (float*)d_out;

  const int N = in_sizes[1] / 3;        // 20000
  const int H = in_sizes[3];            // 256
  const int D = in_sizes[2] / H;        // 128
  const int Npad = (N + 15) & ~15;

  // Carve workspace (256B-aligned segments)
  size_t off = 0;
  auto take = [&](size_t bytes) -> void* {
    void* p = (char*)d_ws + off;
    off += (bytes + 255) & ~(size_t)255;
    return p;
  };
  int*      idx = (int*)      take((size_t)N * KNN * sizeof(int));
  int*      deg = (int*)      take((size_t)N * sizeof(int));
  float*    nrm = (float*)    take((size_t)N * sizeof(float));
  _Float16* wp0 = (_Float16*) take((size_t)D * H * sizeof(_Float16));
  _Float16* wp1 = (_Float16*) take((size_t)H * H * sizeof(_Float16));
  _Float16* wp2 = (_Float16*) take((size_t)H * H * sizeof(_Float16));
  _Float16* agg = (_Float16*) take((size_t)Npad * H * sizeof(_Float16));
  float*    hA  = (float*)    take((size_t)Npad * H * sizeof(float));
  float*    hB  = (float*)    take((size_t)Npad * H * sizeof(float));
  (void)ws_size; (void)n_in; (void)out_size;

  const int B = 256;

  // kNN graph
  knn_kernel<<<cdiv_host(N, B), B, 0, stream>>>(centers, idx, N);

  // degree -> folded norm  (dst_norm = 1/sqrt(16) = 0.25 folded in)
  zero_i32_kernel <<<cdiv_host(N, B), B, 0, stream>>>(deg, N);
  count_deg_kernel<<<cdiv_host(N * KNN, B), B, 0, stream>>>(idx, deg, N * KNN);
  norm_kernel     <<<cdiv_host(N, B), B, 0, stream>>>(deg, nrm, N, 0.25f);

  // pack weights to f16 B-fragment layout
  pack_w_kernel<<<cdiv_host(D * H, B), B, 0, stream>>>(W0, wp0, D, H);
  pack_w_kernel<<<cdiv_host(H * H, B), B, 0, stream>>>(W1, wp1, H, H);
  pack_w_kernel<<<cdiv_host(H * H, B), B, 0, stream>>>(W2, wp2, H, H);

  const int rowTiles = (N + 15) >> 4;
  const int colTiles = H >> 4;
  const int gemmBlocks = cdiv_host(rowTiles * colTiles, B / 32);

  // layer 0: agg(x) -> GEMM(D->H) + relu
  aggregate_kernel<<<cdiv_host(N * (D >> 2), B), B, 0, stream>>>(x, idx, nrm, agg, N, D);
  gemm_wmma_kernel<<<gemmBlocks, B, 0, stream>>>(agg, wp0, b0, hA, N, D, H, 1);

  // layer 1
  aggregate_kernel<<<cdiv_host(N * (H >> 2), B), B, 0, stream>>>(hA, idx, nrm, agg, N, H);
  gemm_wmma_kernel<<<gemmBlocks, B, 0, stream>>>(agg, wp1, b1, hB, N, H, H, 1);

  // layer 2
  aggregate_kernel<<<cdiv_host(N * (H >> 2), B), B, 0, stream>>>(hB, idx, nrm, agg, N, H);
  gemm_wmma_kernel<<<gemmBlocks, B, 0, stream>>>(agg, wp2, b2, hA, N, H, H, 1);

  // classifier head
  final_kernel<<<cdiv_host(N, B), B, 0, stream>>>(hA, Wf, bf, out, N, H);
}
